// StixelHead_62921270887156
// MI455X (gfx1250) — matrix operand order repaired and checked by reading, not compiled, not run on hardware
//
#include <hip/hip_runtime.h>

// ---------------------------------------------------------------------------
// Types for CDNA5 WMMA (gfx1250, wave32)
// ---------------------------------------------------------------------------
typedef __attribute__((ext_vector_type(16))) __bf16 v16bf;
typedef __attribute__((ext_vector_type(8)))  float  v8f;

union ABfrag {
  uint4 u[2];   // two 16-byte chunks = 32 bytes = 16 bf16
  v16bf v;
};

__device__ __forceinline__ unsigned short f2bf(float f) {
  unsigned int u = __float_as_uint(f);
  u += 0x7FFFu + ((u >> 16) & 1u);          // round-to-nearest-even
  return (unsigned short)(u >> 16);
}
__device__ __forceinline__ float bf2f(unsigned short h) {
  return __uint_as_float(((unsigned int)h) << 16);
}

// ---------------------------------------------------------------------------
// Problem constants (from reference)
// ---------------------------------------------------------------------------
constexpr int BATCH = 256;
constexpr int W_IN  = 60;
constexpr int C_IN  = 768;
constexpr int M     = BATCH * W_IN;     // 15360 rows
constexpr int NQKV  = 3 * C_IN;         // 2304
constexpr int OUTC  = 192;
constexpr int KCONV = 2 * C_IN;         // 1536
constexpr int WOUT  = 240;

// ---------------------------------------------------------------------------
// Kernel 0: pack/convert everything to bf16 matrices
//   Xb   [M x 768]      Xb[(b*60+w)*768 + c] = x[b,c,0,w]
//   Wqkv [2304 x 768]   rows: Wq(768) | Wk(768) | Wv(768)  (row-major n,k)
//   Wcb  [192 x 1536]   conv_w row-major
//   biasc[2304] fp32    bq | bk | bv
// ---------------------------------------------------------------------------
__global__ void __launch_bounds__(256)
pack_kernel(const float* __restrict__ x,
            const float* __restrict__ Wq, const float* __restrict__ Wk,
            const float* __restrict__ Wv,
            const float* __restrict__ bq, const float* __restrict__ bk,
            const float* __restrict__ bv,
            const float* __restrict__ convw,
            unsigned short* __restrict__ Xb,
            unsigned short* __restrict__ Wqkv,
            unsigned short* __restrict__ Wcb,
            float* __restrict__ biasc)
{
  const size_t i = (size_t)blockIdx.x * blockDim.x + threadIdx.x;
  if (i < (size_t)M * C_IN) {
    int m = (int)(i / C_IN), c = (int)(i % C_IN);
    int b = m / W_IN, w = m % W_IN;
    Xb[i] = f2bf(x[((size_t)b * C_IN + c) * W_IN + w]);
  }
  if (i < (size_t)NQKV * C_IN) {
    int n = (int)(i / C_IN), k = (int)(i % C_IN);
    float v;
    if (n < C_IN)          v = Wq[(size_t)n * C_IN + k];
    else if (n < 2 * C_IN) v = Wk[(size_t)(n - C_IN) * C_IN + k];
    else                   v = Wv[(size_t)(n - 2 * C_IN) * C_IN + k];
    Wqkv[i] = f2bf(v);
  }
  if (i < (size_t)OUTC * KCONV) Wcb[i] = f2bf(convw[i]);
  if (i < (size_t)NQKV) {
    float v = (i < 768) ? bq[i] : (i < 1536 ? bk[i - 768] : bv[i - 1536]);
    biasc[i] = v;
  }
}

// ---------------------------------------------------------------------------
// Kernel 1: QKV GEMM via v_wmma_f32_16x16x32_bf16
//   QKV[m][n] = sum_k Xb[m][k] * Wqkv[n][k] + biasc[n]     (stored bf16)
//   block = 128 thr = 4 waves; block tile 128M x 64N; wave tile 32M x 64N
//   8 accumulators/wave, double-buffered (ping-pong) fragments so
//   global_load_b128 of step t+1 overlaps the v_wmma of step t.
// ---------------------------------------------------------------------------
__global__ void __launch_bounds__(128)
qkv_gemm_kernel(const unsigned short* __restrict__ Xb,
                const unsigned short* __restrict__ Wqkv,
                const float* __restrict__ biasc,
                unsigned short* __restrict__ QKV)
{
  const int wave = threadIdx.x >> 5;
  const int lane = threadIdx.x & 31;
  const int lm = lane & 15, lh = lane >> 4;
  const int m0 = blockIdx.x * 128 + wave * 32;
  const int n0 = blockIdx.y * 64;

  v8f acc[2][4] = {};

  // per-lane base pointers (K offset added per step)
  const unsigned short* arow[2];
  arow[0] = Xb + (size_t)(m0 + lm) * C_IN;
  arow[1] = Xb + (size_t)(m0 + 16 + lm) * C_IN;
  const unsigned short* brow[4];
#pragma unroll
  for (int j = 0; j < 4; ++j)
    brow[j] = Wqkv + (size_t)(n0 + j * 16 + lm) * C_IN + lh * 16;

  ABfrag aF[2][2], bF[2][4];

  auto loadStep = [&](int kt, ABfrag (&a)[2], ABfrag (&b)[4]) {
#pragma unroll
    for (int t = 0; t < 2; ++t) {
      a[t].u[0] = *(const uint4*)(arow[t] + kt + lh * 16);
      a[t].u[1] = *(const uint4*)(arow[t] + kt + 8 + lh * 16);
    }
#pragma unroll
    for (int j = 0; j < 4; ++j) {
      b[j].u[0] = ((const uint4*)(brow[j] + kt))[0];
      b[j].u[1] = ((const uint4*)(brow[j] + kt))[1];
    }
  };

  loadStep(0, aF[0], bF[0]);
  constexpr int KSTEPS = C_IN / 32;   // 24
#pragma unroll
  for (int it = 0; it < KSTEPS; ++it) {
    const int cur = it & 1, nxt = cur ^ 1;
    if (it + 1 < KSTEPS) {
      loadStep((it + 1) * 32, aF[nxt], bF[nxt]);
      __builtin_prefetch(arow[0] + (it + 3) * 32, 0, 3);
    }
#pragma unroll
    for (int j = 0; j < 4; ++j) {
      acc[0][j] = __builtin_amdgcn_wmma_f32_16x16x32_bf16(
          false, aF[cur][0].v, false, bF[cur][j].v, (short)0, acc[0][j], false, false);
      acc[1][j] = __builtin_amdgcn_wmma_f32_16x16x32_bf16(
          false, aF[cur][1].v, false, bF[cur][j].v, (short)0, acc[1][j], false, false);
    }
  }

  // D layout: VGPR i, lanes 0-15 -> M=i, lanes 16-31 -> M=8+i; N = lane&15
#pragma unroll
  for (int t = 0; t < 2; ++t)
#pragma unroll
    for (int j = 0; j < 4; ++j)
#pragma unroll
      for (int i = 0; i < 8; ++i) {
        int m = m0 + t * 16 + i + 8 * lh;
        int n = n0 + j * 16 + lm;
        QKV[(size_t)m * NQKV + n] = f2bf(acc[t][j][i] + biasc[n]);
      }
}

// ---------------------------------------------------------------------------
// Kernel 2: 3-tap local attention along W (fp32 VALU, wave32 shuffles)
//   one wave per position m = b*60+w ; 4 waves per block
// ---------------------------------------------------------------------------
__global__ void __launch_bounds__(128)
attn_kernel(const unsigned short* __restrict__ QKV,
            unsigned short* __restrict__ ATTN)
{
  const int wave = threadIdx.x >> 5;
  const int lane = threadIdx.x & 31;
  const int m = blockIdx.x * 4 + wave;
  const int w = m % W_IN;

  const unsigned short* qp = QKV + (size_t)m * NQKV;
  const unsigned short* kp = qp + 768;
  const unsigned short* vp = qp + 1536;

  float s0 = 0.f, s1 = 0.f, s2 = 0.f;
  for (int c = lane; c < C_IN; c += 32) {
    float qc = bf2f(qp[c]);
    if (w > 0)        s0 += qc * bf2f(kp[c - NQKV]);   // neighbor w-1
    s1 += qc * bf2f(kp[c]);                            // self
    if (w < W_IN - 1) s2 += qc * bf2f(kp[c + NQKV]);   // neighbor w+1
  }
#pragma unroll
  for (int off = 16; off > 0; off >>= 1) {
    s0 += __shfl_xor(s0, off);
    s1 += __shfl_xor(s1, off);
    s2 += __shfl_xor(s2, off);
  }
  const float inv = 0.03608439182435161f;  // 1/sqrt(768)
  float a0 = (w > 0)        ? s0 * inv : -3.0e38f;
  float a1 = s1 * inv;
  float a2 = (w < W_IN - 1) ? s2 * inv : -3.0e38f;
  float mx = fmaxf(a1, fmaxf(a0, a2));
  float e0 = (w > 0)        ? __expf(a0 - mx) : 0.f;
  float e1 = __expf(a1 - mx);
  float e2 = (w < W_IN - 1) ? __expf(a2 - mx) : 0.f;
  float r = 1.f / (e0 + e1 + e2);
  e0 *= r; e1 *= r; e2 *= r;

  unsigned short* op = ATTN + (size_t)m * C_IN;
  for (int c = lane; c < C_IN; c += 32) {
    float acc = e1 * bf2f(vp[c]);
    if (w > 0)        acc += e0 * bf2f(vp[c - NQKV]);
    if (w < W_IN - 1) acc += e2 * bf2f(vp[c + NQKV]);
    op[c] = f2bf(acc);
  }
}

// ---------------------------------------------------------------------------
// Kernel 3: conv GEMM (N=192, K=1536) + LayerNorm + sigmoid + 4x upsample
//   block = 128 thr = 4 waves; block handles 16 rows x full 192 channels
//   wave j covers channels [j*48, j*48+48); double-buffered fragments.
//   A operand reads ATTN for k<768 and Xb for k>=768 (no materialized concat).
// ---------------------------------------------------------------------------
__global__ void __launch_bounds__(128)
conv_ln_kernel(const unsigned short* __restrict__ ATTN,
               const unsigned short* __restrict__ Xb,
               const unsigned short* __restrict__ Wcb,
               const float* __restrict__ conv_b,
               const float* __restrict__ ln_w,
               const float* __restrict__ ln_b,
               float* __restrict__ out)
{
  __shared__ float ytile[16][OUTC];
  __shared__ float red[16][8][2];
  __shared__ float mu[16], rs[16];

  const int wave = threadIdx.x >> 5;
  const int lane = threadIdx.x & 31;
  const int lm = lane & 15, lh = lane >> 4;
  const int m0 = blockIdx.x * 16;
  const int n0 = wave * 48;

  v8f acc[3] = {};
  const unsigned short* arowA = ATTN + (size_t)(m0 + lm) * C_IN;
  const unsigned short* arowX = Xb   + (size_t)(m0 + lm) * C_IN;
  const unsigned short* brow[3];
#pragma unroll
  for (int j = 0; j < 3; ++j)
    brow[j] = Wcb + (size_t)(n0 + j * 16 + lm) * KCONV + lh * 16;

  ABfrag aF[2], bF[2][3];

  auto loadStep = [&](int kt, ABfrag& a, ABfrag (&b)[3]) {
    const unsigned short* asrc = (kt < C_IN) ? (arowA + kt) : (arowX + (kt - C_IN));
    a.u[0] = *(const uint4*)(asrc + lh * 16);
    a.u[1] = *(const uint4*)(asrc + 8 + lh * 16);
#pragma unroll
    for (int j = 0; j < 3; ++j) {
      b[j].u[0] = ((const uint4*)(brow[j] + kt))[0];
      b[j].u[1] = ((const uint4*)(brow[j] + kt))[1];
    }
  };

  loadStep(0, aF[0], bF[0]);
  constexpr int KSTEPS = KCONV / 32;  // 48
#pragma unroll
  for (int it = 0; it < KSTEPS; ++it) {
    const int cur = it & 1, nxt = cur ^ 1;
    if (it + 1 < KSTEPS) loadStep((it + 1) * 32, aF[nxt], bF[nxt]);
#pragma unroll
    for (int j = 0; j < 3; ++j)
      acc[j] = __builtin_amdgcn_wmma_f32_16x16x32_bf16(
          false, aF[cur].v, false, bF[cur][j].v, (short)0, acc[j], false, false);
  }

#pragma unroll
  for (int j = 0; j < 3; ++j)
#pragma unroll
    for (int i = 0; i < 8; ++i) {
      int n = n0 + j * 16 + lm;
      ytile[i + 8 * lh][n] = acc[j][i] + conv_b[n];
    }
  __syncthreads();

  // LayerNorm over 192 channels per row: 8 partial reducers per row
  const int r0 = threadIdx.x & 15, p = threadIdx.x >> 4;
  float s = 0.f, s2 = 0.f;
  for (int c = p * 24; c < p * 24 + 24; ++c) {
    float v = ytile[r0][c];
    s += v; s2 += v * v;
  }
  red[r0][p][0] = s; red[r0][p][1] = s2;
  __syncthreads();
  if (threadIdx.x < 16) {
    float S = 0.f, S2 = 0.f;
    for (int pp = 0; pp < 8; ++pp) { S += red[threadIdx.x][pp][0]; S2 += red[threadIdx.x][pp][1]; }
    float mean = S * (1.f / OUTC);
    float var  = S2 * (1.f / OUTC) - mean * mean;
    mu[threadIdx.x] = mean;
    rs[threadIdx.x] = rsqrtf(var + 1e-6f);
  }
  __syncthreads();

  // normalize + sigmoid + write (B,3,64,240); upsample = 4x column replication
  const int m = m0 + r0, b = m / W_IN, w = m % W_IN;
  for (int c = p * 24; c < p * 24 + 24; ++c) {
    float val = (ytile[r0][c] - mu[r0]) * rs[r0] * ln_w[c] + ln_b[c];
    float sg = 1.f / (1.f + __expf(-val));
    int g = c >> 6, jj = c & 63;
    float* o = out + ((((size_t)b * 3 + g) * 64 + jj) * WOUT + 4 * w);
    o[0] = sg; o[1] = sg; o[2] = sg; o[3] = sg;
  }
}

// ---------------------------------------------------------------------------
// Host-side launcher
// ---------------------------------------------------------------------------
extern "C" void kernel_launch(void* const* d_in, const int* in_sizes, int n_in,
                              void* d_out, int out_size, void* d_ws, size_t ws_size,
                              hipStream_t stream) {
  const float* x      = (const float*)d_in[0];
  const float* Wq     = (const float*)d_in[1];
  const float* bq     = (const float*)d_in[2];
  const float* Wk     = (const float*)d_in[3];
  const float* bk     = (const float*)d_in[4];
  const float* Wv     = (const float*)d_in[5];
  const float* bv     = (const float*)d_in[6];
  const float* conv_w = (const float*)d_in[7];
  const float* conv_b = (const float*)d_in[8];
  const float* ln_w   = (const float*)d_in[9];
  const float* ln_b   = (const float*)d_in[10];
  float* out          = (float*)d_out;

  // workspace carve (all offsets 16B-aligned)
  char* ws = (char*)d_ws;
  unsigned short* Xb    = (unsigned short*)(ws);                       // 23,592,960 B
  unsigned short* Wqkv  = (unsigned short*)(ws + 23592960);            //  3,538,944 B
  unsigned short* Wcb   = (unsigned short*)(ws + 27131904);            //    589,824 B
  float*          biasc = (float*)        (ws + 27721728);             //      9,216 B
  unsigned short* QKV   = (unsigned short*)(ws + 27730944);            // 70,778,880 B
  unsigned short* ATTN  = (unsigned short*)(ws + 98509824);            // 23,592,960 B
  // total ~122.1 MB

  // K0: pack/convert
  {
    int total = M * C_IN;                         // largest array drives the grid
    pack_kernel<<<(total + 255) / 256, 256, 0, stream>>>(
        x, Wq, Wk, Wv, bq, bk, bv, conv_w, Xb, Wqkv, Wcb, biasc);
  }
  // K1: QKV GEMM  (grid: M/128 x NQKV/64)
  qkv_gemm_kernel<<<dim3(M / 128, NQKV / 64), 128, 0, stream>>>(Xb, Wqkv, biasc, QKV);
  // K2: local attention (4 positions per block)
  attn_kernel<<<M / 4, 128, 0, stream>>>(QKV, ATTN);
  // K3: conv GEMM + LayerNorm + sigmoid + upsample
  conv_ln_kernel<<<M / 16, 128, 0, stream>>>(ATTN, Xb, Wcb, conv_b, ln_w, ln_b, out);
}
